// MultiHeadAttention_8976481648926
// MI455X (gfx1250) — matrix operand order repaired and checked by reading, not compiled, not run on hardware
//
#include <hip/hip_runtime.h>

// Problem sizes (fixed by the reference)
constexpr int Bn  = 4;
constexpr int Tn  = 2048;
constexpr int Cn  = 1024;
constexpr int Hn  = 16;
constexpr int DHn = 64;
constexpr int HDn = Hn * DHn; // 1024

typedef __bf16 v16bf __attribute__((ext_vector_type(16)));
typedef float  v8f   __attribute__((ext_vector_type(8)));
typedef unsigned u32x4 __attribute__((ext_vector_type(4)));
typedef unsigned u32x8 __attribute__((ext_vector_type(8)));

__device__ __forceinline__ v8f wmma_bf16(const unsigned* a, const unsigned* b, v8f c) {
    union U { unsigned u[8]; v16bf v; } A, B;
#pragma unroll
    for (int i = 0; i < 8; ++i) { A.u[i] = a[i]; B.u[i] = b[i]; }
    return __builtin_amdgcn_wmma_f32_16x16x32_bf16(false, A.v, false, B.v,
                                                   (short)0, c, false, false);
}

__device__ __forceinline__ unsigned short f2bf_bits(float f) {
    union { __bf16 h; unsigned short s; } u;
    u.h = (__bf16)f;
    return u.s;
}

// A-fragment (16x32 bf16) pair index for VGPR v, lane-half 'half'
__device__ __forceinline__ int apair(int v, int half) {
    return (v & 3) + 4 * half + 8 * (v >> 2);
}

// ---------------------------------------------------------------- Tensor Data Mover
// 2D tile load: 64 x 64 bf16 tile, rows contiguous (tile_dim0 = 64 elements),
// row stride = stride_d0 elements. D# per CDNA5 ISA.
__device__ __forceinline__ void tdm_load_2d(unsigned lds_off, const void* gaddr,
                                            unsigned stride_d0) {
    unsigned long long ga = (unsigned long long)gaddr;
    u32x4 g0;
    g0[0] = 1u;                                   // count=1 (valid user descriptor)
    g0[1] = lds_off;                              // lds_addr (bytes)
    g0[2] = (unsigned)ga;                         // global_addr[31:0]
    g0[3] = ((unsigned)(ga >> 32) & 0x01FFFFFFu)  // global_addr[56:32]
            | (2u << 30);                         // type = 2 ("image")
    u32x8 g1;
    g1[0] = (1u << 16);                           // data_size=1 (2-byte elements)
    g1[1] = (64u & 0xFFFFu) << 16;                // tensor_dim0[15:0] = 64
    g1[2] = (64u & 0xFFFFu) << 16;                // dim0 hi=0 | tensor_dim1[15:0]=64
    g1[3] = (64u << 16);                          // dim1 hi=0 | tile_dim0=64
    g1[4] = 64u;                                  // tile_dim1=64, tile_dim2=0
    g1[5] = stride_d0;                            // tensor_dim0_stride[31:0]
    g1[6] = 0u;                                   // stride hi / dim1_stride lo
    g1[7] = 0u;
    asm volatile("tensor_load_to_lds %0, %1" :: "s"(g0), "s"(g1) : "memory");
}

// ---------------------------------------------------------------- converts (+ transposes)
__global__ void cvt_bf16_kernel(const float* __restrict__ s, __bf16* __restrict__ d, int n) {
    int i = blockIdx.x * blockDim.x + threadIdx.x;
    int stride = gridDim.x * blockDim.x;
    for (; i < n; i += stride) d[i] = (__bf16)s[i];
}

// (H, C, DH) f32 -> (H, DH, C) bf16   (so GEMM B-tiles are contiguous k-pairs)
__global__ void cvt_wqkv_kernel(const float* __restrict__ s, __bf16* __restrict__ d) {
    int i = blockIdx.x * blockDim.x + threadIdx.x;
    int stride = gridDim.x * blockDim.x;
    const int n = Hn * Cn * DHn;
    for (; i < n; i += stride) {
        int h = i / (DHn * Cn);
        int rem = i - h * (DHn * Cn);
        int dd = rem / Cn;
        int c  = rem - dd * Cn;
        d[i] = (__bf16)s[((size_t)h * Cn + c) * DHn + dd];
    }
}

// (HD, C) f32 -> (C, HD) bf16
__global__ void cvt_wp_kernel(const float* __restrict__ s, __bf16* __restrict__ d) {
    int i = blockIdx.x * blockDim.x + threadIdx.x;
    int stride = gridDim.x * blockDim.x;
    const int n = HDn * Cn;
    for (; i < n; i += stride) {
        int c = i / HDn;
        int k = i - c * HDn;
        d[i] = (__bf16)s[(size_t)k * Cn + c];
    }
}

// ---------------------------------------------------------------- QKV projection
// grid: (B*T/64, H), block 256. q,k in (B,H,T,DH); v in (B,H,DH,T) bf16.
__global__ __launch_bounds__(256) void qkv_gemm_kernel(
    const __bf16* __restrict__ xb,
    const __bf16* __restrict__ wq, const __bf16* __restrict__ wk, const __bf16* __restrict__ wv,
    __bf16* __restrict__ qo, __bf16* __restrict__ ko, __bf16* __restrict__ vo)
{
    __shared__ unsigned Xt[64][16];      // 64 rows x 16 k-pairs
    __shared__ unsigned Wt[3][64][16];   // [n][k-pair] (weights pre-transposed)
    const int tid  = threadIdx.x;
    const int wi   = tid >> 5;
    const int lane = tid & 31;
    const int half = lane >> 4, ln = lane & 15;
    const int r0   = blockIdx.x * 64;    // global row in [0, B*T)
    const int h    = blockIdx.y;

    v8f acc[6];
#pragma unroll
    for (int j = 0; j < 6; ++j) acc[j] = {};

    for (int kc = 0; kc < Cn; kc += 32) {
#pragma unroll
        for (int i = 0; i < 4; ++i) {
            int idx = tid + i * 256;
            int row = idx >> 4, kp = idx & 15;
            Xt[row][kp] = *(const unsigned*)(xb + (size_t)(r0 + row) * Cn + kc + 2 * kp);
        }
#pragma unroll
        for (int m = 0; m < 3; ++m) {
            const __bf16* W = ((m == 0) ? wq : (m == 1) ? wk : wv)
                              + (size_t)h * DHn * Cn;   // (DH, C) layout
#pragma unroll
            for (int i = 0; i < 4; ++i) {
                int idx = tid + i * 256;
                int n = idx >> 4, kp = idx & 15;
                Wt[m][n][kp] = *(const unsigned*)(W + (size_t)n * Cn + kc + 2 * kp);
            }
        }
        __syncthreads();
#pragma unroll
        for (int j = 0; j < 6; ++j) {
            int g = wi + 8 * j;            // 48 jobs: matrix x rsub x csub
            int m  = g >> 4;
            int rs = (g >> 2) & 3;
            int cs = g & 3;
            unsigned afrag[8], bfrag[8];
#pragma unroll
            for (int v = 0; v < 8; ++v) {
                afrag[v] = Xt[rs * 16 + ln][apair(v, half)];
                bfrag[v] = Wt[m][cs * 16 + ln][apair(v, half)];
            }
            acc[j] = wmma_bf16(afrag, bfrag, acc[j]);
        }
        __syncthreads();
    }
#pragma unroll
    for (int j = 0; j < 6; ++j) {
        int g = wi + 8 * j;
        int m = g >> 4, rs = (g >> 2) & 3, cs = g & 3;
#pragma unroll
        for (int r = 0; r < 8; ++r) {
            int row = r0 + rs * 16 + r + 8 * half;
            int b = row >> 11;            // / Tn
            int t = row & (Tn - 1);
            int bh = b * Hn + h;
            __bf16 val = (__bf16)acc[j][r];
            if (m == 0)
                qo[((size_t)bh * Tn + t) * DHn + cs * 16 + ln] = val;
            else if (m == 1)
                ko[((size_t)bh * Tn + t) * DHn + cs * 16 + ln] = val;
            else  // V transposed: (B,H,DH,T)
                vo[((size_t)bh * DHn + cs * 16 + ln) * Tn + t] = val;
        }
    }
}

// ---------------------------------------------------------------- flash attention
// grid: (B*H, T/128), block 256 (8 waves, each owns a 16-row query subtile).
// K/V tiles fetched by the Tensor Data Mover with double buffering.
__global__ __launch_bounds__(256) void attn_kernel(
    const __bf16* __restrict__ qg, const __bf16* __restrict__ kg,
    const __bf16* __restrict__ vg, __bf16* __restrict__ att)
{
    __shared__ unsigned Kt[2][64][32];   // [buf][key][dh-pair]  (B frag for Q*K^T)
    __shared__ unsigned Vt[2][64][32];   // [buf][dh][key-pair]  (B frag for P*V)
    __shared__ unsigned Pb[8][16][32];   // per-wave P tile 16x64 bf16
    const int tid  = threadIdx.x;
    const int wi   = tid >> 5;
    const int lane = tid & 31;
    const int half = lane >> 4, ln = lane & 15;
    const int bh = blockIdx.x;
    const int b = bh >> 4, h = bh & 15;
    const int t0w = blockIdx.y * 128 + wi * 16;     // wave's first query row
    const size_t qkbase = (size_t)bh * Tn * DHn;    // q,k: (B,H,T,DH)
    const size_t vbase  = (size_t)bh * DHn * Tn;    // v:   (B,H,DH,T)

    const unsigned kt_lds = (unsigned)(uintptr_t)&Kt[0][0][0];
    const unsigned vt_lds = (unsigned)(uintptr_t)&Vt[0][0][0];

    // Q fragments (2 K-steps of 32 over DH=64), loaded once
    unsigned qa[2][8];
#pragma unroll
    for (int s = 0; s < 2; ++s)
#pragma unroll
        for (int v = 0; v < 8; ++v)
            qa[s][v] = *(const unsigned*)(qg + qkbase + (size_t)(t0w + ln) * DHn
                                          + s * 32 + 2 * apair(v, half));

    v8f O[4];
#pragma unroll
    for (int c = 0; c < 4; ++c) O[c] = {};
    float mrow[8], lrow[8];
#pragma unroll
    for (int r = 0; r < 8; ++r) { mrow[r] = -3.0e38f; lrow[r] = 0.0f; }
    const float scale = 0.03125f;  // 1/sqrt(C) = 1/32

    const int nkb = blockIdx.y * 2 + 2;   // causal: only blocks up to this tile

    // TDM prefetch of tile 0 into buffer 0 (wave 0 drives the DMA)
    if (wi == 0) {
        tdm_load_2d(kt_lds, kg + qkbase, DHn);       // 64 keys x 64 dh, contiguous rows
        tdm_load_2d(vt_lds, vg + vbase, Tn);         // 64 dh x 64 keys, row stride T
    }

    for (int kb = 0; kb < nkb; ++kb) {
        const int kt0 = kb * 64;
        const int cur = kb & 1;
        if (wi == 0) {
            if (kb + 1 < nkb) {
                const int nt0 = (kb + 1) * 64;
                const unsigned boff = (unsigned)(((kb + 1) & 1) * 64 * 32 * 4);
                tdm_load_2d(kt_lds + boff, kg + qkbase + (size_t)nt0 * DHn, DHn);
                tdm_load_2d(vt_lds + boff, vg + vbase + nt0, Tn);
                __builtin_amdgcn_s_wait_tensorcnt(2);  // current tile done, next in flight
            } else {
                __builtin_amdgcn_s_wait_tensorcnt(0);
            }
        }
        __syncthreads();
        if (kt0 <= t0w + 15) {
            // ---- scores S = Q K^T for 4 key subtiles
            float sv[4][8];
#pragma unroll
            for (int ns = 0; ns < 4; ++ns) {
                v8f sacc = {};
#pragma unroll
                for (int s = 0; s < 2; ++s) {
                    unsigned bfrag[8];
#pragma unroll
                    for (int v = 0; v < 8; ++v)
                        bfrag[v] = Kt[cur][ns * 16 + ln][s * 16 + apair(v, half)];
                    sacc = wmma_bf16(qa[s], bfrag, sacc);
                }
                int keyi = kt0 + ns * 16 + ln;
#pragma unroll
                for (int r = 0; r < 8; ++r) {
                    int qi = t0w + r + 8 * half;
                    float x = sacc[r] * scale;
                    sv[ns][r] = (keyi <= qi) ? x : -3.0e38f;
                }
            }
            // ---- online softmax update
#pragma unroll
            for (int r = 0; r < 8; ++r) {
                float mx = fmaxf(fmaxf(sv[0][r], sv[1][r]), fmaxf(sv[2][r], sv[3][r]));
                for (int d = 1; d < 16; d <<= 1) mx = fmaxf(mx, __shfl_xor(mx, d, 32));
                float mn = fmaxf(mrow[r], mx);
                float alpha = __expf(mrow[r] - mn);
                mrow[r] = mn;
                float rs = 0.0f;
#pragma unroll
                for (int ns = 0; ns < 4; ++ns) {
                    float p = __expf(sv[ns][r] - mn);
                    sv[ns][r] = p;
                    rs += p;
                }
                for (int d = 1; d < 16; d <<= 1) rs += __shfl_xor(rs, d, 32);
                lrow[r] = lrow[r] * alpha + rs;
#pragma unroll
                for (int c = 0; c < 4; ++c) O[c][r] *= alpha;
            }
            // ---- restage P (C layout -> A layout) via per-wave LDS
#pragma unroll
            for (int ns = 0; ns < 4; ++ns)
#pragma unroll
                for (int r = 0; r < 8; ++r)
                    ((unsigned short*)&Pb[wi][0][0])[(r + 8 * half) * 64 + ns * 16 + ln]
                        = f2bf_bits(sv[ns][r]);
            unsigned pa[2][8];
#pragma unroll
            for (int s = 0; s < 2; ++s)
#pragma unroll
                for (int v = 0; v < 8; ++v)
                    pa[s][v] = Pb[wi][ln][s * 16 + apair(v, half)];
            // ---- O += P V
#pragma unroll
            for (int c = 0; c < 4; ++c)
#pragma unroll
                for (int s = 0; s < 2; ++s) {
                    unsigned bfrag[8];
#pragma unroll
                    for (int v = 0; v < 8; ++v)
                        bfrag[v] = Vt[cur][c * 16 + ln][s * 16 + apair(v, half)];
                    O[c] = wmma_bf16(pa[s], bfrag, O[c]);
                }
        }
        __syncthreads();
    }
    // finalize: att layout (B, T, H*DH) bf16
#pragma unroll
    for (int r = 0; r < 8; ++r) {
        float inv = 1.0f / lrow[r];
        int t = t0w + r + 8 * half;
        size_t o = ((size_t)b * Tn + t) * HDn + (size_t)h * DHn;
#pragma unroll
        for (int c = 0; c < 4; ++c)
            att[o + c * 16 + ln] = (__bf16)(O[c][r] * inv);
    }
}

// ---------------------------------------------------------------- output projection + bias
// grid: (B*T/64, C/64), block 256. Wp pre-transposed to (C, HD).
__global__ __launch_bounds__(256) void oproj_kernel(
    const __bf16* __restrict__ att, const __bf16* __restrict__ wp,
    const float* __restrict__ bp, float* __restrict__ out)
{
    __shared__ unsigned At[64][16];
    __shared__ unsigned Bt[64][16];
    const int tid  = threadIdx.x;
    const int wi   = tid >> 5;
    const int lane = tid & 31;
    const int half = lane >> 4, ln = lane & 15;
    const int r0 = blockIdx.x * 64;
    const int n0 = blockIdx.y * 64;

    v8f acc[2];
    acc[0] = {}; acc[1] = {};
    for (int kc = 0; kc < HDn; kc += 32) {
#pragma unroll
        for (int i = 0; i < 4; ++i) {
            int idx = tid + i * 256;
            int row = idx >> 4, kp = idx & 15;
            At[row][kp] = *(const unsigned*)(att + (size_t)(r0 + row) * HDn + kc + 2 * kp);
        }
#pragma unroll
        for (int i = 0; i < 4; ++i) {
            int idx = tid + i * 256;
            int n = idx >> 4, kp = idx & 15;
            Bt[n][kp] = *(const unsigned*)(wp + (size_t)(n0 + n) * HDn + kc + 2 * kp);
        }
        __syncthreads();
#pragma unroll
        for (int j = 0; j < 2; ++j) {
            int g = wi + 8 * j;
            int rs = g >> 2, cs = g & 3;
            unsigned afrag[8], bfrag[8];
#pragma unroll
            for (int v = 0; v < 8; ++v) {
                afrag[v] = At[rs * 16 + ln][apair(v, half)];
                bfrag[v] = Bt[cs * 16 + ln][apair(v, half)];
            }
            acc[j] = wmma_bf16(afrag, bfrag, acc[j]);
        }
        __syncthreads();
    }
#pragma unroll
    for (int j = 0; j < 2; ++j) {
        int g = wi + 8 * j;
        int rs = g >> 2, cs = g & 3;
#pragma unroll
        for (int r = 0; r < 8; ++r) {
            int row = r0 + rs * 16 + r + 8 * half;
            int n = n0 + cs * 16 + ln;
            out[(size_t)row * Cn + n] = acc[j][r] + bp[n];
        }
    }
}

// ---------------------------------------------------------------- launcher
extern "C" void kernel_launch(void* const* d_in, const int* in_sizes, int n_in,
                              void* d_out, int out_size, void* d_ws, size_t ws_size,
                              hipStream_t stream) {
    const float* x  = (const float*)d_in[0];
    const float* Wq = (const float*)d_in[1];
    const float* Wk = (const float*)d_in[2];
    const float* Wv = (const float*)d_in[3];
    const float* Wp = (const float*)d_in[4];
    const float* bp = (const float*)d_in[5];
    float* out = (float*)d_out;

    char* p = (char*)d_ws;
    auto carve = [&](size_t elems) -> __bf16* {
        __bf16* r = (__bf16*)p;
        p += (elems * sizeof(__bf16) + 255) & ~(size_t)255;
        return r;
    };
    const size_t nX = (size_t)Bn * Tn * Cn;
    const size_t nW = (size_t)Hn * Cn * DHn;
    const size_t nP = (size_t)HDn * Cn;
    const size_t nQ = (size_t)Bn * Hn * Tn * DHn;
    __bf16* xb  = carve(nX);
    __bf16* wqb = carve(nW);   // (H, DH, C)
    __bf16* wkb = carve(nW);
    __bf16* wvb = carve(nW);
    __bf16* wpb = carve(nP);   // (C, HD)
    __bf16* qb  = carve(nQ);   // (B,H,T,DH)
    __bf16* kb  = carve(nQ);   // (B,H,T,DH)
    __bf16* vb  = carve(nQ);   // (B,H,DH,T)
    __bf16* attb = carve((size_t)Bn * Tn * HDn);

    cvt_bf16_kernel<<<1024, 256, 0, stream>>>(x, xb, (int)nX);
    cvt_wqkv_kernel<<<512, 256, 0, stream>>>(Wq, wqb);
    cvt_wqkv_kernel<<<512, 256, 0, stream>>>(Wk, wkb);
    cvt_wqkv_kernel<<<512, 256, 0, stream>>>(Wv, wvb);
    cvt_wp_kernel<<<512, 256, 0, stream>>>(Wp, wpb);

    qkv_gemm_kernel<<<dim3((Bn * Tn) / 64, Hn), 256, 0, stream>>>(
        xb, wqb, wkb, wvb, qb, kb, vb);

    attn_kernel<<<dim3(Bn * Hn, Tn / 128), 256, 0, stream>>>(qb, kb, vb, attb);

    oproj_kernel<<<dim3((Bn * Tn) / 64, Cn / 64), 256, 0, stream>>>(attb, wpb, bp, out);
}
